// GCN_LSTM_8065948582442
// MI455X (gfx1250) — matrix operand order repaired and checked by reading, not compiled, not run on hardware
//
#include <hip/hip_runtime.h>

// ---------------------------------------------------------------------------
// GCN-LSTM for MI455X (gfx1250).  All GEMMs run on V_WMMA_F32_16X16X32_BF16,
// global->LDS staging uses the CDNA5 async-to-LDS path when available.
// T=12, B=8, N=1000 (padded to 1024), IN=128, H=256, gates=4H=1024.
// ---------------------------------------------------------------------------

#define TT 12
#define BB 8
#define NN 1000
#define NP 1024
#define HH 256
#define G4 1024          // 4*H
#define LNCNT (NN * HH)  // elements per (b,gate) layer-norm group

#define MT 128           // macro tile M
#define NT 128           // macro tile N
#define KT 32            // K step (one bf16 WMMA depth)
#define LROW 20          // dwords per LDS row: 16 data + 4 pad (keeps 16B align)
#define BUFDW ((MT + NT) * LROW)

typedef __attribute__((ext_vector_type(16))) __bf16 v16bf;
typedef __attribute__((ext_vector_type(8)))  float  v8f;
typedef int v4i __attribute__((vector_size(16)));   // matches builtin param type

union Frag { unsigned int u[8]; v16bf v; };

__device__ __forceinline__ unsigned short f2bf(float f) {
    union { float f; unsigned int u; } x; x.f = f;
    unsigned int u = x.u;
    u += 0x7FFFu + ((u >> 16) & 1u);      // round-to-nearest-even
    return (unsigned short)(u >> 16);
}

// ---- CDNA5 async global->LDS copy (ASYNCcnt tracked), with safe fallback ----
#if defined(__has_builtin)
#if __has_builtin(__builtin_amdgcn_global_load_async_to_lds_b128) && \
    __has_builtin(__builtin_amdgcn_s_wait_asynccnt)
#define USE_ASYNC 1
#endif
#endif
#ifndef USE_ASYNC
#define USE_ASYNC 0
#endif

#if USE_ASYNC
__device__ __forceinline__ void async_ld16(const unsigned short* g, unsigned int* l) {
    __builtin_amdgcn_global_load_async_to_lds_b128(
        (__attribute__((address_space(1))) v4i*)g,
        (__attribute__((address_space(3))) v4i*)l, 0, 0);
}
#endif

// ---------------------------------------------------------------------------
// Batched bf16 GEMM:  C[b] = A[b] (1024 x K) * B[b] (K x 1024)  (+bias)
//   A  : row-major  [m][K]            (lda = K row stride)
//   Bt : k-major    [ncol][K]         (ldb = K row stride)  == B transposed
//   out: either Cf  f32 row-major [m][1024] (+bias)          (GEMM-2)
//         or   Ct  bf16 TRANSPOSED [ncol][1024 m]            (GEMM-1 -> feeds
//                                                             next GEMM's Bt)
// Block: 256 threads = 8 waves; macro tile 128x128; wave tile 32x64
// (2x4 accumulators -> 8 WMMAs per 32-deep K step).
// ---------------------------------------------------------------------------
__global__ __launch_bounds__(256)
void gemm_bf16_wmma(const unsigned short* __restrict__ A, long long strideAb, int lda,
                    const unsigned short* __restrict__ Bt, long long strideBb, int ldb,
                    float* __restrict__ Cf, unsigned short* __restrict__ Ct,
                    long long strideCb, int ldc,
                    const float* __restrict__ bias, int K)
{
    __shared__ __align__(16) unsigned int smem[2 * BUFDW];   // 40 KB, double buffer

    const int tid  = threadIdx.x;
    const int w    = tid >> 5;
    const int lane = tid & 31;
    const int wm   = w & 3;              // 4 M sub-tiles of 32
    const int wn   = w >> 2;             // 2 N sub-tiles of 64
    const int lm   = lane & 15;
    const int lh   = lane >> 4;

    const int m0 = blockIdx.y * MT;
    const int n0 = blockIdx.x * NT;
    const int b  = blockIdx.z;

    const unsigned short* Ab  = A  + (long long)b * strideAb;
    const unsigned short* Btb = Bt + (long long)b * strideBb;

    // per-thread 16B chunk coordinates (2 chunks per operand per tile)
    const int ar = tid >> 2;             // row 0..63 (and +64)
    const int ac = tid & 3;              // 16B chunk within 64B k-strip
    const unsigned short* gA0 = Ab  + (long long)(m0 + ar) * lda + ac * 8;
    const unsigned short* gA1 = gA0 + (long long)64 * lda;
    const unsigned short* gB0 = Btb + (long long)(n0 + ar) * ldb + ac * 8;
    const unsigned short* gB1 = gB0 + (long long)64 * ldb;
    const int lA0 = ar * LROW + ac * 4;
    const int lA1 = (64 + ar) * LROW + ac * 4;
    const int lB0 = (MT + ar) * LROW + ac * 4;
    const int lB1 = (MT + 64 + ar) * LROW + ac * 4;

    v8f acc[2][4];
    #pragma unroll
    for (int i = 0; i < 2; ++i)
        #pragma unroll
        for (int j = 0; j < 4; ++j) acc[i][j] = {};

    int cur = 0;

#if USE_ASYNC
    {   // prologue: async-load tile 0 into buffer 0
        unsigned int* bf0 = smem;
        async_ld16(gA0, bf0 + lA0);
        async_ld16(gA1, bf0 + lA1);
        async_ld16(gB0, bf0 + lB0);
        async_ld16(gB1, bf0 + lB1);
        __builtin_amdgcn_s_wait_asynccnt(0);
        __syncthreads();
    }
#else
    {   // prologue: register-staged tile 0
        uint4 rA0 = *(const uint4*)gA0, rA1 = *(const uint4*)gA1;
        uint4 rB0 = *(const uint4*)gB0, rB1 = *(const uint4*)gB1;
        unsigned int* bf0 = smem;
        *(uint4*)(bf0 + lA0) = rA0; *(uint4*)(bf0 + lA1) = rA1;
        *(uint4*)(bf0 + lB0) = rB0; *(uint4*)(bf0 + lB1) = rB1;
        __syncthreads();
    }
#endif

    for (int k0 = 0; k0 < K; k0 += KT) {
        const bool next = (k0 + KT) < K;
        unsigned int* bufc = smem + cur * BUFDW;
        unsigned int* bufn = smem + (cur ^ 1) * BUFDW;

#if USE_ASYNC
        if (next) {                       // async-fill the other buffer
            async_ld16(gA0 + k0 + KT, bufn + lA0);
            async_ld16(gA1 + k0 + KT, bufn + lA1);
            async_ld16(gB0 + k0 + KT, bufn + lB0);
            async_ld16(gB1 + k0 + KT, bufn + lB1);
        }
#else
        uint4 rA0, rA1, rB0, rB1;
        if (next) {                       // issue loads early, hide behind WMMAs
            rA0 = *(const uint4*)(gA0 + k0 + KT);
            rA1 = *(const uint4*)(gA1 + k0 + KT);
            rB0 = *(const uint4*)(gB0 + k0 + KT);
            rB1 = *(const uint4*)(gB1 + k0 + KT);
            __builtin_prefetch((const void*)(gA0 + k0 + 2 * KT), 0, 1);
            __builtin_prefetch((const void*)(gB0 + k0 + 2 * KT), 0, 1);
        }
#endif

        // ---- fragments per ISA 7.12.2 bf16 layouts (k-pairs packed in LDS) ----
        const unsigned int* As = bufc;
        const unsigned int* Bs = bufc + MT * LROW;
        Frag a[2], bf[4];
        #pragma unroll
        for (int v = 0; v < 8; ++v) {
            const int dA = ((v >> 2) << 3) + (lh << 2) + (v & 3);
            a[0].u[v] = As[(wm * 32 + lm)      * LROW + dA];
            a[1].u[v] = As[(wm * 32 + 16 + lm) * LROW + dA];
            const int dB = ((v >> 2) << 3) + (lh << 2) + (v & 3);
            bf[0].u[v] = Bs[(wn * 64 + lm)      * LROW + dB];
            bf[1].u[v] = Bs[(wn * 64 + 16 + lm) * LROW + dB];
            bf[2].u[v] = Bs[(wn * 64 + 32 + lm) * LROW + dB];
            bf[3].u[v] = Bs[(wn * 64 + 48 + lm) * LROW + dB];
        }
        #pragma unroll
        for (int j = 0; j < 4; ++j) {
            acc[0][j] = __builtin_amdgcn_wmma_f32_16x16x32_bf16(
                false, a[0].v, false, bf[j].v, (short)0, acc[0][j], false, false);
            acc[1][j] = __builtin_amdgcn_wmma_f32_16x16x32_bf16(
                false, a[1].v, false, bf[j].v, (short)0, acc[1][j], false, false);
        }

#if USE_ASYNC
        __builtin_amdgcn_s_wait_asynccnt(0);
#else
        if (next) {
            *(uint4*)(bufn + lA0) = rA0; *(uint4*)(bufn + lA1) = rA1;
            *(uint4*)(bufn + lB0) = rB0; *(uint4*)(bufn + lB1) = rB1;
        }
#endif
        __syncthreads();
        cur ^= 1;
    }

    // ------------------------------- epilogue -------------------------------
    if (Cf) {
        // f32 row-major output (+bias): GEMM-2
        float* C = Cf + (long long)b * strideCb;
        #pragma unroll
        for (int j = 0; j < 4; ++j) {
            const int gn = n0 + wn * 64 + j * 16 + lm;
            const float bv = bias ? bias[gn] : 0.0f;
            #pragma unroll
            for (int im = 0; im < 2; ++im)
                #pragma unroll
                for (int r = 0; r < 8; ++r) {
                    const int gm = m0 + wm * 32 + im * 16 + lh * 8 + r;
                    C[(long long)gm * ldc + gn] = acc[im][j][r] + bv;
                }
        }
    } else {
        // bf16 TRANSPOSED output via LDS staging (coalesced): GEMM-1
        unsigned short* ts = (unsigned short*)smem;
        const int TS = MT + 8;            // 136: padded row, keeps 16B chunks aligned
        #pragma unroll
        for (int im = 0; im < 2; ++im)
            #pragma unroll
            for (int j = 0; j < 4; ++j)
                #pragma unroll
                for (int r = 0; r < 8; ++r) {
                    const int lmr = wm * 32 + im * 16 + lh * 8 + r;   // local m
                    const int lnr = wn * 64 + j * 16 + lm;            // local n
                    ts[lnr * TS + lmr] = f2bf(acc[im][j][r]);
                }
        __syncthreads();
        unsigned short* C = Ct + (long long)b * strideCb;
        #pragma unroll
        for (int i = 0; i < 8; ++i) {
            const int id = tid + i * 256;     // 2048 x 16B chunks
            const int rr = id >> 4;           // ncol row 0..127
            const int cc = id & 15;           // 16B chunk along m
            uint4 d = *(const uint4*)(ts + rr * TS + cc * 8);
            *(uint4*)(C + (long long)(n0 + rr) * ldc + m0 + cc * 8) = d;
        }
    }
}

// ---------------------------------------------------------------------------
// Helper kernels
// ---------------------------------------------------------------------------
__global__ void zero_f32(float* p, int n) {
    int i = blockIdx.x * 256 + threadIdx.x;
    if (i < n) p[i] = 0.0f;
}

// stacked gate weights [Wx ; Wh] -> bf16 K-MAJOR: wt[g][k], k in [0,K1+K2)
__global__ void pack_wt(unsigned short* dst, const float* wx, int K1,
                        const float* wh, int K2) {
    int Kc = K1 + K2;
    int idx = blockIdx.x * 256 + threadIdx.x;
    if (idx >= G4 * Kc) return;
    int k = idx % Kc, g = idx / Kc;
    float v = (k < K1) ? wx[k * G4 + g] : wh[(k - K1) * G4 + g];
    dst[idx] = f2bf(v);
}

// adjacency -> bf16, zero-padded to 1024x1024 (padding annihilates junk rows)
__global__ void conv_adj(unsigned short* dst, const float* adj) {
    int idx = blockIdx.x * 256 + threadIdx.x;
    if (idx >= NP * NP) return;
    int r = idx >> 10, c = idx & (NP - 1);
    float v = (r < NN && c < NN) ? adj[r * NN + c] : 0.0f;
    dst[idx] = f2bf(v);
}

// Acat[b][n][0:K1]=s1, [K1:K1+K2]=s2 (bf16 row-major, batch stride 1024*Kc)
__global__ void pack_acat(unsigned short* dst, int Kc,
                          const float* __restrict__ s1, int K1,
                          const float* __restrict__ s2, int K2) {
    int idx = blockIdx.x * 256 + threadIdx.x;
    if (idx >= BB * NN * Kc) return;
    int k  = idx % Kc;
    int rn = idx / Kc;
    int n  = rn % NN, b = rn / NN;
    float v = (k < K1) ? s1[((long long)b * NN + n) * K1 + k]
                       : s2[((long long)b * NN + n) * K2 + (k - K1)];
    dst[((long long)b * NP + n) * Kc + k] = f2bf(v);
}

// LayerNorm stats over (N,H) per (batch, gate): 32 blocks
__global__ __launch_bounds__(256)
void ln_stats(const float* __restrict__ pre, float* __restrict__ stats) {
    int bg = blockIdx.x;
    int b = bg >> 2, g = bg & 3;
    const float* base = pre + (long long)b * (NP * G4) + g * HH;
    float s = 0.0f, ss = 0.0f;
    for (int idx = threadIdx.x; idx < LNCNT; idx += 256) {
        int n = idx >> 8, h = idx & 255;
        float v = base[(long long)n * G4 + h];
        s += v; ss += v * v;
    }
    __shared__ float rs[256], rq[256];
    rs[threadIdx.x] = s; rq[threadIdx.x] = ss;
    __syncthreads();
    for (int o = 128; o > 0; o >>= 1) {
        if (threadIdx.x < o) {
            rs[threadIdx.x] += rs[threadIdx.x + o];
            rq[threadIdx.x] += rq[threadIdx.x + o];
        }
        __syncthreads();
    }
    if (threadIdx.x == 0) {
        float m   = rs[0] * (1.0f / (float)LNCNT);
        float var = rq[0] * (1.0f / (float)LNCNT) - m * m;
        stats[bg * 2 + 0] = m;
        stats[bg * 2 + 1] = rsqrtf(var + 1e-5f);
    }
}

// Fused gate activations + LSTM cell update (and optional output store)
__global__ __launch_bounds__(256)
void cell_update(const float* __restrict__ pre, const float* __restrict__ stats,
                 float* __restrict__ c, float* __restrict__ h,
                 float* __restrict__ out, int t) {
    int idx = blockIdx.x * 256 + threadIdx.x;
    if (idx >= BB * NN * HH) return;
    int b   = idx / (NN * HH);
    int rem = idx % (NN * HH);
    int n   = rem >> 8, hh = rem & 255;
    long long pb = (long long)b * (NP * G4) + (long long)n * G4;

    float mi = stats[(b * 4 + 0) * 2], ri = stats[(b * 4 + 0) * 2 + 1];
    float mf = stats[(b * 4 + 1) * 2], rf = stats[(b * 4 + 1) * 2 + 1];
    float mo = stats[(b * 4 + 2) * 2], ro = stats[(b * 4 + 2) * 2 + 1];
    float mg = stats[(b * 4 + 3) * 2], rg = stats[(b * 4 + 3) * 2 + 1];

    float vi = pre[pb + 0 * HH + hh];
    float vf = pre[pb + 1 * HH + hh];
    float vo = pre[pb + 2 * HH + hh];
    float vg = pre[pb + 3 * HH + hh];

    float ig = 1.0f / (1.0f + expf(-(vi - mi) * ri));
    float fg = 1.0f / (1.0f + expf(-(vf - mf) * rf));
    float og = 1.0f / (1.0f + expf(-(vo - mo) * ro));
    float gg = tanhf((vg - mg) * rg);

    float cn = fg * c[idx] + ig * gg;
    float hn = og * tanhf(cn);
    c[idx] = cn;
    h[idx] = hn;
    if (out) out[((long long)b * TT + t) * (NN * HH) + rem] = hn;
}

// ---------------------------------------------------------------------------
extern "C" void kernel_launch(void* const* d_in, const int* in_sizes, int n_in,
                              void* d_out, int out_size, void* d_ws, size_t ws_size,
                              hipStream_t stream) {
    (void)in_sizes; (void)n_in; (void)out_size; (void)ws_size;
    const float* x   = (const float*)d_in[0];
    const float* adj = (const float*)d_in[1];
    const float* Wx0 = (const float*)d_in[2];
    const float* bx0 = (const float*)d_in[3];
    const float* Wh0 = (const float*)d_in[4];
    const float* Wx1 = (const float*)d_in[5];
    const float* bx1 = (const float*)d_in[6];
    const float* Wh1 = (const float*)d_in[7];
    float* out = (float*)d_out;

    // workspace layout (~96 MB)
    char* ws = (char*)d_ws;
    unsigned short* adjB  = (unsigned short*)ws;  ws += (size_t)NP * NP * 2;        // 2 MB
    unsigned short* w0t   = (unsigned short*)ws;  ws += (size_t)G4 * 384 * 2;       // .75 MB
    unsigned short* w1t   = (unsigned short*)ws;  ws += (size_t)G4 * 512 * 2;       // 1 MB
    unsigned short* acat  = (unsigned short*)ws;  ws += (size_t)BB * NP * 512 * 2;  // 8 MB
    unsigned short* supT  = (unsigned short*)ws;  ws += (size_t)BB * G4 * NP * 2;   // 16 MB
    float* pre            = (float*)ws;           ws += (size_t)BB * NP * G4 * 4;   // 32 MB
    float* h0             = (float*)ws;           ws += (size_t)BB * NN * HH * 4;   // 8 MB
    float* c0             = (float*)ws;           ws += (size_t)BB * NN * HH * 4;
    float* h1             = (float*)ws;           ws += (size_t)BB * NN * HH * 4;
    float* c1             = (float*)ws;           ws += (size_t)BB * NN * HH * 4;
    float* stats          = (float*)ws;           ws += 256;

    const long long sSup = (long long)G4 * NP;   // batch strides (elements)
    const long long sPre = (long long)NP * G4;

    // ---- one-time (per launch) preparation ----
    {
        int nW0 = G4 * 384, nW1 = G4 * 512;
        pack_wt<<<(nW0 + 255) / 256, 256, 0, stream>>>(w0t, Wx0, 128, Wh0, 256);
        pack_wt<<<(nW1 + 255) / 256, 256, 0, stream>>>(w1t, Wx1, 256, Wh1, 256);
        conv_adj<<<(NP * NP + 255) / 256, 256, 0, stream>>>(adjB, adj);
        int nz = 4 * BB * NN * HH;               // h0,c0,h1,c1 are contiguous
        zero_f32<<<(nz + 255) / 256, 256, 0, stream>>>(h0, nz);
    }

    const dim3 gGemm(G4 / NT, NP / MT, BB);      // (8, 8, 8)

    for (int t = 0; t < TT; ++t) {
        // ================= layer 0 =================
        {
            const float* xt = x + (size_t)t * BB * NN * 128;
            int nA = BB * NN * 384;
            pack_acat<<<(nA + 255) / 256, 256, 0, stream>>>(acat, 384, xt, 128, h0, 256);
            // supT = ([x|h0] @ [Wx0;Wh0])^T   (bf16, k-major for next GEMM)
            gemm_bf16_wmma<<<gGemm, 256, 0, stream>>>(
                acat, (long long)NP * 384, 384, w0t, 0, 384,
                nullptr, supT, sSup, NP, nullptr, 384);
            // pre = adj @ support + bx0       (f32)
            gemm_bf16_wmma<<<gGemm, 256, 0, stream>>>(
                adjB, 0, NP, supT, sSup, NP,
                pre, nullptr, sPre, G4, bx0, NP);
            ln_stats<<<32, 256, 0, stream>>>(pre, stats);
            cell_update<<<(BB * NN * HH + 255) / 256, 256, 0, stream>>>(
                pre, stats, c0, h0, nullptr, t);
        }
        // ================= layer 1 =================
        {
            int nA = BB * NN * 512;
            pack_acat<<<(nA + 255) / 256, 256, 0, stream>>>(acat, 512, h0, 256, h1, 256);
            gemm_bf16_wmma<<<gGemm, 256, 0, stream>>>(
                acat, (long long)NP * 512, 512, w1t, 0, 512,
                nullptr, supT, sSup, NP, nullptr, 512);
            gemm_bf16_wmma<<<gGemm, 256, 0, stream>>>(
                adjB, 0, NP, supT, sSup, NP,
                pre, nullptr, sPre, G4, bx1, NP);
            ln_stats<<<32, 256, 0, stream>>>(pre, stats);
            cell_update<<<(BB * NN * HH + 255) / 256, 256, 0, stream>>>(
                pre, stats, c1, h1, out, t);
        }
    }
}
